// Model_2413771620585
// MI455X (gfx1250) — compile-verified, matrix operand order checked by reading, not run on hardware
//
#include <hip/hip_runtime.h>
#include <math.h>

// ---------------- model dims ----------------
constexpr int B  = 512;
constexpr int L  = 100;
constexpr int V  = 100000;
constexpr int D  = 128;
constexpr int NH = 2;
constexpr int NL = 2;
constexpr int DH = D / NH;   // 64
constexpr int FF = 4 * D;    // 512
constexpr int BL = B * L;    // 51200
constexpr float NEGINF = -1e9f;

typedef __attribute__((ext_vector_type(16))) __bf16 v16bf;
typedef __attribute__((ext_vector_type(8)))  __bf16 v8bf;
typedef __attribute__((ext_vector_type(4)))  __bf16 v4bf;
typedef __attribute__((ext_vector_type(8)))  float  v8f;

#define DEV __device__ __forceinline__

DEV void store_bf16x4(__bf16* p, float4 v)
{
    v4bf t;
    t[0] = (__bf16)v.x; t[1] = (__bf16)v.y;
    t[2] = (__bf16)v.z; t[3] = (__bf16)v.w;
    *(v4bf*)p = t;
}

// =====================================================================
// WMMA GEMM: C[M,N] = act( A[M,K] @ W[N,K]^T + bias[N] )
// A, W fp32 in HBM; staged in float4 granularity (global_load_b128,
// clause-batched), converted to bf16 (v_cvt_pk_bf16_f32) into LDS,
// consumed with ds_load_b128 fragment loads + v_wmma_f32_16x16x32_bf16.
// Block tile 128(M) x 64(N), 256 threads = 8 waves, wave tile 32x32.
// Requires: M % 128 == 0, K % 32 == 0.  N edge: B row index is clamped
// (OOB columns only pollute accumulator lanes that are never stored).
// ACT: 0=none, 1=exact GELU, 2=tanh
// =====================================================================
template <int ACT>
__global__ __launch_bounds__(256)
void gemm_wmma(const float* __restrict__ A, const float* __restrict__ W,
               const float* __restrict__ bias, float* __restrict__ C,
               int M, int N, int K)
{
    constexpr int LDT = 40;                 // bf16 row stride (80B: keeps b128 alignment)
    __shared__ __align__(16) __bf16 As[128 * LDT];
    __shared__ __align__(16) __bf16 Bs[64 * LDT];

    const int tid  = threadIdx.x;
    const int lane = tid & 31;
    const int wave = tid >> 5;
    const int wm   = wave >> 1;             // 0..3  (M sub-block of 32)
    const int wn   = wave & 1;              // 0..1  (N sub-block of 32)
    const int rowBase = blockIdx.y * 128;
    const int colBase = blockIdx.x * 64;

    // float4-granularity staging coordinates (invariant across k-steps)
    const int sRow = tid >> 3;              // 0..31, pass p adds 32
    const int sCol = (tid & 7) * 4;         // 0,4,...,28

    v8f zero;
#pragma unroll
    for (int t = 0; t < 8; ++t) zero[t] = 0.0f;
    v8f acc[2][2];
#pragma unroll
    for (int i = 0; i < 2; ++i)
#pragma unroll
        for (int j = 0; j < 2; ++j) acc[i][j] = zero;

    const int half = lane >> 4;             // which 16-lane half
    const int kb   = half * 8;              // K base of this half's fragment
    const int mrow = lane & 15;

    // global base pointers for this thread's staging lanes
    const float* Abase = A + (size_t)(rowBase + sRow) * K + sCol;
    const size_t Astride = (size_t)32 * K;  // +32 rows per pass

    // clamped B row pointers (branch-free edge handling)
    const float* Wbase0 = W + (size_t)min(colBase + sRow,      N - 1) * K + sCol;
    const float* Wbase1 = W + (size_t)min(colBase + sRow + 32, N - 1) * K + sCol;

    for (int kt = 0; kt < K; kt += 32) {
        // ---- issue all global loads for this k-step into registers ----
        float4 av[4];
#pragma unroll
        for (int p = 0; p < 4; ++p)
            av[p] = *(const float4*)(Abase + p * Astride + kt);

        float4 bv[2];
        bv[0] = *(const float4*)(Wbase0 + kt);
        bv[1] = *(const float4*)(Wbase1 + kt);

        // previous iteration's fragment reads must finish before overwrite
        __syncthreads();

        // ---- convert + store to LDS (packed bf16, 8B stores) ----
#pragma unroll
        for (int p = 0; p < 4; ++p)
            store_bf16x4(&As[(sRow + p * 32) * LDT + sCol], av[p]);
#pragma unroll
        for (int p = 0; p < 2; ++p)
            store_bf16x4(&Bs[(sRow + p * 32) * LDT + sCol], bv[p]);
        __syncthreads();

        // ---- build fragments per ISA 16-bit A 16x32 layout ----
        v16bf afrag[2], bfrag[2];
#pragma unroll
        for (int s = 0; s < 2; ++s) {
            int ar = wm * 32 + s * 16 + mrow;
            v8bf alo = *(const v8bf*)&As[ar * LDT + kb];
            v8bf ahi = *(const v8bf*)&As[ar * LDT + kb + 16];
#pragma unroll
            for (int e = 0; e < 8; ++e) { afrag[s][e] = alo[e]; afrag[s][e + 8] = ahi[e]; }
            int br = wn * 32 + s * 16 + mrow;
            v8bf blo = *(const v8bf*)&Bs[br * LDT + kb];
            v8bf bhi = *(const v8bf*)&Bs[br * LDT + kb + 16];
#pragma unroll
            for (int e = 0; e < 8; ++e) { bfrag[s][e] = blo[e]; bfrag[s][e + 8] = bhi[e]; }
        }

#pragma unroll
        for (int i = 0; i < 2; ++i)
#pragma unroll
            for (int j = 0; j < 2; ++j)
                acc[i][j] = __builtin_amdgcn_wmma_f32_16x16x32_bf16(
                    false, afrag[i], false, bfrag[j],
                    (short)0, acc[i][j], false, false);
    }

    // ---- epilogue: C 16x16 layout: VGPR r -> M = r + 8*half, N = lane&15 ----
    const int ncol = lane & 15;
#pragma unroll
    for (int i = 0; i < 2; ++i) {
#pragma unroll
        for (int j = 0; j < 2; ++j) {
            int n = colBase + wn * 32 + j * 16 + ncol;
            if (n >= N) continue;
            float bv = bias ? bias[n] : 0.0f;
#pragma unroll
            for (int r = 0; r < 8; ++r) {
                int m = rowBase + wm * 32 + i * 16 + r + 8 * half;
                float v = acc[i][j][r] + bv;
                if (ACT == 1) v = 0.5f * v * (1.0f + erff(v * 0.70710678118f));
                else if (ACT == 2) v = tanhf(v);
                C[(size_t)m * N + n] = v;
            }
        }
    }
}

// =====================================================================
// Attention: one wave per (b, h, q). qkv layout [BL, 3D] (q | k | v).
// =====================================================================
__global__ __launch_bounds__(128)
void attn_kernel(const float* __restrict__ qkv, const int* __restrict__ x,
                 float* __restrict__ o)
{
    int gw = blockIdx.x * 4 + (threadIdx.x >> 5);   // exact: B*NH*L / 4 blocks
    int lane = threadIdx.x & 31;
    int q = gw % L;
    int h = (gw / L) % NH;
    int b = gw / (L * NH);
    const float scale = 0.125f;                      // 1/sqrt(64)

    const size_t rowQ = (size_t)(b * L + q) * (3 * D) + h * DH;
    float q0 = qkv[rowQ + 2 * lane];
    float q1 = qkv[rowQ + 2 * lane + 1];

    float sc[4];
#pragma unroll
    for (int j = 0; j < 4; ++j) sc[j] = NEGINF;

    for (int k = 0; k < L; ++k) {
        const size_t rowK = (size_t)(b * L + k) * (3 * D) + D + h * DH;
        float p = q0 * qkv[rowK + 2 * lane] + q1 * qkv[rowK + 2 * lane + 1];
#pragma unroll
        for (int m = 16; m > 0; m >>= 1) p += __shfl_xor(p, m, 32);
        float s = p * scale;
        bool masked = (k > q) || (x[b * L + k] == 0);
        s = masked ? NEGINF : s;
        if ((k & 31) == lane) sc[k >> 5] = s;
    }
    float mx = sc[0];
#pragma unroll
    for (int j = 1; j < 4; ++j) mx = fmaxf(mx, sc[j]);
#pragma unroll
    for (int m = 16; m > 0; m >>= 1) mx = fmaxf(mx, __shfl_xor(mx, m, 32));
    float e[4];
    float sum = 0.0f;
#pragma unroll
    for (int j = 0; j < 4; ++j) {
        int k = j * 32 + lane;
        e[j] = (k < L) ? __expf(sc[j] - mx) : 0.0f;
        sum += e[j];
    }
#pragma unroll
    for (int m = 16; m > 0; m >>= 1) sum += __shfl_xor(sum, m, 32);
    float inv = 1.0f / sum;

    float o0 = 0.0f, o1 = 0.0f;
#pragma unroll
    for (int j = 0; j < 4; ++j) {
        int kmax = (j * 32 + 32 <= L) ? 32 : (L - j * 32);
        for (int kk = 0; kk < kmax; ++kk) {
            float a = __shfl(e[j], kk, 32) * inv;
            const size_t rowV = (size_t)(b * L + j * 32 + kk) * (3 * D) + 2 * D + h * DH;
            o0 += a * qkv[rowV + 2 * lane];
            o1 += a * qkv[rowV + 2 * lane + 1];
        }
    }
    size_t rowO = (size_t)(b * L + q) * D + h * DH;
    o[rowO + 2 * lane]     = o0;
    o[rowO + 2 * lane + 1] = o1;
}

// =====================================================================
// LayerNorm with optional residual: Y[row] = LN(X[row] + R[row]) * g + b
// =====================================================================
__global__ __launch_bounds__(128)
void ln_kernel(const float* __restrict__ X, const float* __restrict__ R,
               const float* __restrict__ g, const float* __restrict__ bt,
               float* __restrict__ Y)
{
    __shared__ float red[128];
    int row = blockIdx.x, d = threadIdx.x;
    float v = X[(size_t)row * D + d] + (R ? R[(size_t)row * D + d] : 0.0f);
    red[d] = v; __syncthreads();
    for (int s = 64; s > 0; s >>= 1) { if (d < s) red[d] += red[d + s]; __syncthreads(); }
    float mean = red[0] * (1.0f / D); __syncthreads();
    float c = v - mean;
    red[d] = c * c; __syncthreads();
    for (int s = 64; s > 0; s >>= 1) { if (d < s) red[d] += red[d + s]; __syncthreads(); }
    float var = red[0] * (1.0f / D);
    Y[(size_t)row * D + d] = c * rsqrtf(var + 1e-5f) * g[d] + bt[d];
}

// =====================================================================
__global__ void embed_kernel(const int* __restrict__ x, const float* __restrict__ emb,
                             const float* __restrict__ pos, float* __restrict__ h)
{
    int bl = blockIdx.x, d = threadIdx.x;
    int it = x[bl];
    h[(size_t)bl * D + d] = emb[(size_t)it * D + d] + pos[(bl % L) * D + d];
}

__global__ __launch_bounds__(128)
void hT_kernel(const int* __restrict__ mask, const float* __restrict__ h,
               float* __restrict__ hT)
{
    __shared__ int red[128];
    int b = blockIdx.x, d = threadIdx.x;
    red[d] = (d < L) ? mask[b * L + d] : 0;
    __syncthreads();
    for (int s = 64; s > 0; s >>= 1) { if (d < s) red[d] += red[d + s]; __syncthreads(); }
    int last = max(red[0] - 1, 0);
    hT[(size_t)b * D + d] = h[((size_t)b * L + last) * D + d];
}

// =====================================================================
// Local-graph preprocessing: sorted-unique vals, compressed seq, edges,
// row counts, n, index of last item in vals.  One block per sample.
// =====================================================================
DEV int lower_bound_s(const int* a, int key)
{
    int lo = 0, hi = L;
    while (lo < hi) { int mid = (lo + hi) >> 1; if (a[mid] < key) lo = mid + 1; else hi = mid; }
    return lo;
}

__global__ __launch_bounds__(128)
void local_prep_kernel(const int* __restrict__ x, int* __restrict__ valsG,
                       int* __restrict__ uG, int* __restrict__ wG,
                       int* __restrict__ cntG, int* __restrict__ nG,
                       int* __restrict__ lastIdxG)
{
    __shared__ int seq[L], ss[L], tmpv[L], vals[L], comp[L], cnt[L];
    __shared__ int nsh;
    int b = blockIdx.x, t = threadIdx.x;
    if (t < L) { seq[t] = x[b * L + t]; comp[t] = 0; cnt[t] = 0; }
    __syncthreads();
    if (t < L) {                                   // stable rank sort of seq
        int v = seq[t], r = 0;
        for (int j = 0; j < L; ++j) { int sj = seq[j]; r += (sj < v) || (sj == v && j < t); }
        ss[r] = v;
    }
    __syncthreads();
    if (t < L) tmpv[t] = (t == 0 || ss[t] != ss[t - 1]) ? ss[t] : V;
    __syncthreads();
    if (t < L) {                                   // re-sort (duplicates -> sentinel V)
        int v = tmpv[t], r = 0;
        for (int j = 0; j < L; ++j) { int sj = tmpv[j]; r += (sj < v) || (sj == v && j < t); }
        vals[r] = v;
    }
    if (t == 0) {
        int n = 0;
        for (int j = 0; j < L; ++j) n += (seq[j] != 0);
        nsh = n;
    }
    if (t < L && seq[t] != 0) {                    // stable compaction of nonzeros
        int pos = 0;
        for (int j = 0; j < t; ++j) pos += (seq[j] != 0);
        comp[pos] = seq[t];
    }
    __syncthreads();
    int n = nsh;
    if (t < L - 1) {
        int u = -1, w = -1;
        if (t < n - 1) {
            u = lower_bound_s(vals, comp[t]);
            w = lower_bound_s(vals, comp[t + 1]);
            atomicAdd(&cnt[u], 1);
        }
        uG[b * L + t] = u;
        wG[b * L + t] = w;
    }
    __syncthreads();
    if (t < L) { valsG[b * L + t] = vals[t]; cntG[b * L + t] = cnt[t]; }
    if (t == 0) {
        nG[b] = n;
        lastIdxG[b] = lower_bound_s(vals, comp[max(n - 1, 0)]);
    }
}

__global__ void h0_kernel(const int* __restrict__ valsG, const float* __restrict__ emb,
                          float* __restrict__ H0)
{
    int bl = blockIdx.x, d = threadIdx.x;
    int v = valsG[bl];
    bool valid = (v != 0) && (v != V);
    H0[(size_t)bl * D + d] = emb[(size_t)(valid ? v : 0) * D + d];
}

// xg[b,i,:] += (sum_{e: u[e]==i} Hout[b, w[e], :]) / (cnt[b,i] + 1e-8)
__global__ void xg_add_kernel(const int* __restrict__ uG, const int* __restrict__ wG,
                              const int* __restrict__ cntG, const float* __restrict__ Hout,
                              float* __restrict__ xg)
{
    int bl = blockIdx.x;
    int b = bl / L, i = bl % L, d = threadIdx.x;
    const int* ub = uG + b * L;
    const int* wb = wG + b * L;
    float acc = 0.0f;
    for (int e = 0; e < L - 1; ++e)
        if (ub[e] == i) acc += Hout[((size_t)b * L + wb[e]) * D + d];
    float c = (float)cntG[b * L + i];
    xg[(size_t)bl * D + d] += acc / (c + 1e-8f);
}

__global__ void gru_kernel(const float* __restrict__ gi, const float* __restrict__ gh,
                           const float* __restrict__ H0, float* __restrict__ H)
{
    int bl = blockIdx.x, d = threadIdx.x;
    size_t o3 = (size_t)bl * (3 * D);
    float ir = gi[o3 + d], iz = gi[o3 + D + d], ig = gi[o3 + 2 * D + d];
    float hr = gh[o3 + d], hz = gh[o3 + D + d], hg = gh[o3 + 2 * D + d];
    float r = 1.0f / (1.0f + __expf(-(ir + hr)));
    float z = 1.0f / (1.0f + __expf(-(iz + hz)));
    float g = tanhf(ig + r * hg);
    H[(size_t)bl * D + d] = (1.0f - z) * g + z * H0[(size_t)bl * D + d];
}

__global__ __launch_bounds__(128)
void readout_kernel(const float* __restrict__ H, const int* __restrict__ valsG,
                    const int* __restrict__ nG, const int* __restrict__ lastIdxG,
                    float* __restrict__ loc)
{
    __shared__ float hl[D];
    __shared__ float att[128];
    __shared__ float red[128];
    int b = blockIdx.x, t = threadIdx.x;
    int n = nG[b];
    hl[t] = H[((size_t)b * L + lastIdxG[b]) * D + t];
    __syncthreads();
    float s = NEGINF;
    if (t < L) {
        int v = valsG[b * L + t];
        if (v != 0 && v != V) {
            const float* Hi = H + ((size_t)b * L + t) * D;
            float acc = 0.0f;
            for (int d = 0; d < D; ++d) acc += Hi[d] * hl[d];
            s = acc;
        }
    }
    att[t] = s; red[t] = s;
    __syncthreads();
    for (int st = 64; st > 0; st >>= 1) { if (t < st) red[t] = fmaxf(red[t], red[t + st]); __syncthreads(); }
    float mx = red[0]; __syncthreads();
    float e = (t < L) ? __expf(att[t] - mx) : 0.0f;
    red[t] = e; __syncthreads();
    for (int st = 64; st > 0; st >>= 1) { if (t < st) red[t] += red[t + st]; __syncthreads(); }
    float inv = 1.0f / red[0]; __syncthreads();
    att[t] = e * inv;
    __syncthreads();
    float out = 0.0f;
    for (int i = 0; i < L; ++i) out += att[i] * H[((size_t)b * L + i) * D + t];
    loc[(size_t)b * D + t] = (n > 0) ? out : 0.0f;
}

__global__ void concat_kernel(const float* __restrict__ hT, const float* __restrict__ loc,
                              float* __restrict__ cat)
{
    int b = blockIdx.x, d = threadIdx.x;
    cat[(size_t)b * 2 * D + d]     = hT[(size_t)b * D + d];
    cat[(size_t)b * 2 * D + D + d] = loc[(size_t)b * D + d];
}

// =====================================================================
extern "C" void kernel_launch(void* const* d_in, const int* in_sizes, int n_in,
                              void* d_out, int out_size, void* d_ws, size_t ws_size,
                              hipStream_t stream)
{
    const int*   x        = (const int*)d_in[0];
    const int*   amask    = (const int*)d_in[1];
    const float* item_emb = (const float*)d_in[2];
    const float* pos_emb  = (const float*)d_in[3];

    struct Layer {
        const float *in_w, *in_b, *out_w, *out_b;
        const float *ln1_g, *ln1_b, *ln2_g, *ln2_b;
        const float *w1, *b1, *w2, *b2;
    } lay[NL];
    int idx = 4;
    for (int l = 0; l < NL; ++l) {
        lay[l].in_w  = (const float*)d_in[idx++];
        lay[l].in_b  = (const float*)d_in[idx++];
        lay[l].out_w = (const float*)d_in[idx++];
        lay[l].out_b = (const float*)d_in[idx++];
        lay[l].ln1_g = (const float*)d_in[idx++];
        lay[l].ln1_b = (const float*)d_in[idx++];
        lay[l].ln2_g = (const float*)d_in[idx++];
        lay[l].ln2_b = (const float*)d_in[idx++];
        lay[l].w1    = (const float*)d_in[idx++];
        lay[l].b1    = (const float*)d_in[idx++];
        lay[l].w2    = (const float*)d_in[idx++];
        lay[l].b2    = (const float*)d_in[idx++];
    }
    const float* norm_g   = (const float*)d_in[idx++];
    const float* norm_b   = (const float*)d_in[idx++];
    const float* W_in     = (const float*)d_in[idx++];
    const float* W_out    = (const float*)d_in[idx++];
    const float* gru_w_ih = (const float*)d_in[idx++];
    const float* gru_w_hh = (const float*)d_in[idx++];
    const float* gru_b_ih = (const float*)d_in[idx++];
    const float* gru_b_hh = (const float*)d_in[idx++];
    const float* fuse_w   = (const float*)d_in[idx++];
    const float* fuse_b   = (const float*)d_in[idx++];

    // ---------------- workspace layout (floats) ----------------
    float* ws = (float*)d_ws;
    size_t off = 0;
    float* h      = ws + off; off += (size_t)BL * D;       // hidden / later H0
    float* big    = ws + off; off += (size_t)BL * FF;      // qkv / f1 / xg+gi
    float* obuf   = ws + off; off += (size_t)BL * D;       // attn-out / f2 / Hout
    float* ghb    = ws + off; off += (size_t)BL * 3 * D;   // GRU gh
    float* Hgru   = ws + off; off += (size_t)BL * D;       // attn-proj temp / GRU H
    float* hT     = ws + off; off += (size_t)B * D;
    float* locb   = ws + off; off += (size_t)B * D;
    float* catb   = ws + off; off += (size_t)B * 2 * D;
    float* fusedb = ws + off; off += (size_t)B * D;
    int* ib = (int*)(ws + off);
    int* valsG    = ib; ib += BL;
    int* uG       = ib; ib += BL;
    int* wG       = ib; ib += BL;
    int* cntG     = ib; ib += BL;
    int* nG       = ib; ib += B;
    int* lastIdxG = ib; ib += B;

    float* xg = big;                       // [BL, D]
    float* gi = big + (size_t)BL * D;      // [BL, 3D]

    const dim3 blk(256);

    // ---------------- transformer ----------------
    embed_kernel<<<BL, D, 0, stream>>>(x, item_emb, pos_emb, h);
    for (int l = 0; l < NL; ++l) {
        gemm_wmma<0><<<dim3((3 * D) / 64, BL / 128), blk, 0, stream>>>(
            h, lay[l].in_w, lay[l].in_b, big, BL, 3 * D, D);
        attn_kernel<<<(B * NH * L) / 4, 128, 0, stream>>>(big, x, obuf);
        gemm_wmma<0><<<dim3(D / 64, BL / 128), blk, 0, stream>>>(
            obuf, lay[l].out_w, lay[l].out_b, Hgru, BL, D, D);
        ln_kernel<<<BL, 128, 0, stream>>>(h, Hgru, lay[l].ln1_g, lay[l].ln1_b, h);
        gemm_wmma<1><<<dim3(FF / 64, BL / 128), blk, 0, stream>>>(
            h, lay[l].w1, lay[l].b1, big, BL, FF, D);                    // GELU
        gemm_wmma<0><<<dim3(D / 64, BL / 128), blk, 0, stream>>>(
            big, lay[l].w2, lay[l].b2, obuf, BL, D, FF);
        ln_kernel<<<BL, 128, 0, stream>>>(h, obuf, lay[l].ln2_g, lay[l].ln2_b, h);
    }
    ln_kernel<<<BL, 128, 0, stream>>>(h, nullptr, norm_g, norm_b, h);
    hT_kernel<<<B, 128, 0, stream>>>(amask, h, hT);

    // ---------------- local graph ----------------
    local_prep_kernel<<<B, 128, 0, stream>>>(x, valsG, uG, wG, cntG, nG, lastIdxG);
    h0_kernel<<<BL, D, 0, stream>>>(valsG, item_emb, h);                 // H0 -> h
    gemm_wmma<0><<<dim3(D / 64, BL / 128), blk, 0, stream>>>(
        h, W_out, nullptr, obuf, BL, D, D);                              // Hout
    gemm_wmma<0><<<dim3(D / 64, BL / 128), blk, 0, stream>>>(
        h, W_in, nullptr, xg, BL, D, D);                                 // Hin -> xg
    xg_add_kernel<<<BL, D, 0, stream>>>(uG, wG, cntG, obuf, xg);         // xg += A@Hout
    gemm_wmma<0><<<dim3((3 * D) / 64, BL / 128), blk, 0, stream>>>(
        xg, gru_w_ih, gru_b_ih, gi, BL, 3 * D, D);                       // gi
    gemm_wmma<0><<<dim3((3 * D) / 64, BL / 128), blk, 0, stream>>>(
        h, gru_w_hh, gru_b_hh, ghb, BL, 3 * D, D);                       // gh
    gru_kernel<<<BL, D, 0, stream>>>(gi, ghb, h, Hgru);                  // H
    readout_kernel<<<B, 128, 0, stream>>>(Hgru, valsG, nG, lastIdxG, locb);

    // ---------------- fuse + logits ----------------
    concat_kernel<<<B, D, 0, stream>>>(hT, locb, catb);
    gemm_wmma<2><<<dim3(D / 64, B / 128), blk, 0, stream>>>(
        catb, fuse_w, fuse_b, fusedb, B, D, 2 * D);                      // tanh
    gemm_wmma<0><<<dim3((V + 63) / 64, B / 128), blk, 0, stream>>>(
        fusedb, item_emb, nullptr, (float*)d_out, B, V, D);              // logits
}
